// GAT_7078106104193
// MI455X (gfx1250) — matrix operand order repaired and checked by reading, compiled, not run on hardware
//
#include <hip/hip_runtime.h>
#include <cstdint>
#include <cstddef>

#define N_NODES 4096
#define N_FEATC 512
#define N_HID   64
#define N_HEADS 8
#define N_CLASS 16
#define ALPHA_L 0.2f

typedef __attribute__((ext_vector_type(16))) __bf16    v16bf;
typedef __attribute__((ext_vector_type(8)))  float     v8f;
typedef __attribute__((ext_vector_type(4)))  unsigned  v4u;
typedef __attribute__((ext_vector_type(4)))  int       v4i;
typedef __attribute__((ext_vector_type(4)))  float     v4f;

union BF16Vec { v16bf v; unsigned short s[16]; v4u q[2]; };
union I16u    { v4i q[4]; int   s[16]; };
union F16u    { v4f q[4]; float s[16]; };

__device__ __forceinline__ unsigned short f2bf(float x) {
  union { float f; unsigned u; } c; c.f = x;
  unsigned u = c.u;
  unsigned r = u + 0x7FFFu + ((u >> 16) & 1u);   // round-to-nearest-even
  return (unsigned short)(r >> 16);
}

// ---------------------------------------------------------------- converts
__global__ void cvt_x_bf16(const float* __restrict__ in,
                           unsigned short* __restrict__ out, int n) {
  int i = blockIdx.x * 256 + threadIdx.x;
  if (i < n) out[i] = f2bf(in[i]);
}

// W1 [H][F][HID] -> W1T [(h*HID+o)][F]  (B stored N-major so K is contiguous)
__global__ void cvt_w1_bf16(const float* __restrict__ W1,
                            unsigned short* __restrict__ W1T) {
  int t = blockIdx.x * 256 + threadIdx.x;
  if (t >= N_HEADS * N_FEATC * N_HID) return;
  int h = t / (N_FEATC * N_HID);
  int r = t - h * (N_FEATC * N_HID);
  int f = r / N_HID;
  int o = r - f * N_HID;
  W1T[(size_t)(h * N_HID + o) * N_FEATC + f] = f2bf(W1[t]);
}

// W2 [F][C] -> W2T [C][F]
__global__ void cvt_w2_bf16(const float* __restrict__ W2,
                            unsigned short* __restrict__ W2T) {
  int t = blockIdx.x * 256 + threadIdx.x;
  if (t >= N_FEATC * N_CLASS) return;
  int k = t / N_CLASS;
  int c = t - k * N_CLASS;
  W2T[(size_t)c * N_FEATC + k] = f2bf(W2[t]);
}

// ---------------------------------------------------------------- WMMA GEMM
// C[M,ldc] = A[M,512] @ B[512,ldc], A row-major bf16, B given transposed [N][512].
// Block = 8 waves; wave -> 16 x (NT*16) tile. Also writes CT bf16 [N][4096].
template<int NT>
__global__ __launch_bounds__(256) void gemm_bf16(
    const unsigned short* __restrict__ A,
    const unsigned short* __restrict__ BT,
    float* __restrict__ C,
    unsigned short* __restrict__ CT,
    int ldc)
{
  const int wave = threadIdx.x >> 5;
  const int lane = threadIdx.x & 31;
  const int hi = lane >> 4, lo = lane & 15;
  const int Mbase = blockIdx.x * 128 + wave * 16;
  const int Nbase = blockIdx.y * (NT * 16);

  v8f acc[NT] = {};

  const unsigned short* aRow = A + (size_t)(Mbase + lo) * N_FEATC;
  for (int k0 = 0; k0 < N_FEATC; k0 += 32) {
    BF16Vec a;
    const unsigned short* ap = aRow + k0 + hi * 8;      // K = kb..kb+7, kb+16..kb+23
    a.q[0] = *(const v4u*)(ap);
    a.q[1] = *(const v4u*)(ap + 16);
#pragma unroll
    for (int n = 0; n < NT; ++n) {
      BF16Vec b;
      const unsigned short* bp =
          BT + (size_t)(Nbase + n * 16 + lo) * N_FEATC + k0 + hi * 8;
      b.q[0] = *(const v4u*)(bp);
      b.q[1] = *(const v4u*)(bp + 16);
      acc[n] = __builtin_amdgcn_wmma_f32_16x16x32_bf16(
          false, a.v, false, b.v, (short)0, acc[n], false, false);
    }
  }

#pragma unroll
  for (int n = 0; n < NT; ++n)
#pragma unroll
    for (int r = 0; r < 8; ++r) {
      int row = Mbase + r + (hi << 3);
      int col = Nbase + n * 16 + lo;
      float v = acc[n][r];
      C[(size_t)row * ldc + col] = v;
      CT[(size_t)col * N_NODES + row] = f2bf(v);
    }
}

// ------------------------------------------------------- attention f-vectors
template<int NH, int HID>
__global__ void fdot_kernel(const float* __restrict__ Wh,
                            const float* __restrict__ a,
                            float* __restrict__ fsrc,
                            float* __restrict__ fdst)
{
  int t = blockIdx.x * 256 + threadIdx.x;
  if (t >= NH * N_NODES) return;
  int h = t / N_NODES;
  int i = t - h * N_NODES;
  const float* w  = Wh + (size_t)i * (NH * HID) + h * HID;
  const float* av = a + h * 2 * HID;
  float s0 = 0.f, s1 = 0.f;
#pragma unroll 4
  for (int o = 0; o < HID; ++o) {
    s0 = fmaf(w[o], av[o], s0);
    s1 = fmaf(w[o], av[HID + o], s1);
  }
  fsrc[h * N_NODES + i] = s0;
  fdst[h * N_NODES + i] = s1;
}

// --------------------------------------- masked-softmax row stats (max, sum)
// m_i = leakyrelu(f_src_i + max_{adj} f_dst_j); s_i = sum_{adj} exp(e - m_i)
template<int NH>
__global__ __launch_bounds__(256) void row_stats(
    const int* __restrict__ adj,
    const float* __restrict__ fsrc, const float* __restrict__ fdst,
    float* __restrict__ mOut, float* __restrict__ sOut)
{
  __shared__ float red[256];
  __shared__ float mh[NH];
  const int i = blockIdx.x;
  const int tid = threadIdx.x;
  const int* arow = adj + (size_t)i * N_NODES;

  float lmax[NH];
#pragma unroll
  for (int h = 0; h < NH; ++h) lmax[h] = -3.0e38f;
  for (int j = tid; j < N_NODES; j += 256) {
    if (arow[j] > 0) {
#pragma unroll
      for (int h = 0; h < NH; ++h)
        lmax[h] = fmaxf(lmax[h], fdst[h * N_NODES + j]);
    }
  }
#pragma unroll
  for (int h = 0; h < NH; ++h) {
    red[tid] = lmax[h]; __syncthreads();
    for (int s = 128; s > 0; s >>= 1) {
      if (tid < s) red[tid] = fmaxf(red[tid], red[tid + s]);
      __syncthreads();
    }
    if (tid == 0) {
      float e = fsrc[h * N_NODES + i] + red[0];
      mh[h] = e > 0.f ? e : ALPHA_L * e;
    }
    __syncthreads();
  }

  float lsum[NH], fs[NH];
#pragma unroll
  for (int h = 0; h < NH; ++h) { lsum[h] = 0.f; fs[h] = fsrc[h * N_NODES + i]; }
  for (int j = tid; j < N_NODES; j += 256) {
    if (arow[j] > 0) {
#pragma unroll
      for (int h = 0; h < NH; ++h) {
        float e = fs[h] + fdst[h * N_NODES + j];
        e = e > 0.f ? e : ALPHA_L * e;
        lsum[h] += __expf(e - mh[h]);
      }
    }
  }
#pragma unroll
  for (int h = 0; h < NH; ++h) {
    red[tid] = lsum[h]; __syncthreads();
    for (int s = 128; s > 0; s >>= 1) {
      if (tid < s) red[tid] += red[tid + s];
      __syncthreads();
    }
    if (tid == 0) { mOut[h * N_NODES + i] = mh[h]; sOut[h * N_NODES + i] = red[0]; }
    __syncthreads();
  }
}

// ---------------------------------------------- fused att-build + WMMA matmul
// out[i, head*NT*16 + o] = sum_j att(i,j) * Wh[j, o] ; att built in-register,
// already normalized by 1/s so no rescale pass is needed.
template<int NT, bool DO_ELU, bool F32OUT>
__global__ __launch_bounds__(256) void att_mm(
    const int* __restrict__ adj,
    const float* __restrict__ fsrc, const float* __restrict__ fdst,
    const float* __restrict__ mIn,  const float* __restrict__ sIn,
    const unsigned short* __restrict__ WhT,   // [cols][N_NODES] bf16
    unsigned short* __restrict__ outBF,       // [N_NODES][ldOut] bf16
    unsigned short* __restrict__ outBFT,      // [ldOut][N_NODES] bf16
    float* __restrict__ outF,                 // [N_NODES][NT*16] f32
    int ldOut)
{
  const int wave = threadIdx.x >> 5;
  const int lane = threadIdx.x & 31;
  const int hi = lane >> 4, lo = lane & 15;
  const int head = blockIdx.y;
  const int Mbase = blockIdx.x * 128 + wave * 16;
  const int i = Mbase + lo;
  const float fs = fsrc[head * N_NODES + i];
  const float mi = mIn[head * N_NODES + i];
  const float rs = 1.0f / sIn[head * N_NODES + i];
  const int* arow = adj + (size_t)i * N_NODES;
  const float* frow = fdst + head * N_NODES;
  const int colHead = head * (NT * 16);

  v8f acc[NT] = {};

  for (int k0 = 0; k0 < N_NODES; k0 += 32) {
    const int jA = k0 + hi * 8;
    I16u av; F16u fv;
    av.q[0] = *(const v4i*)(arow + jA);
    av.q[1] = *(const v4i*)(arow + jA + 4);
    av.q[2] = *(const v4i*)(arow + jA + 16);
    av.q[3] = *(const v4i*)(arow + jA + 20);
    fv.q[0] = *(const v4f*)(frow + jA);
    fv.q[1] = *(const v4f*)(frow + jA + 4);
    fv.q[2] = *(const v4f*)(frow + jA + 16);
    fv.q[3] = *(const v4f*)(frow + jA + 20);
    __builtin_prefetch(arow + jA + 32, 0, 0);   // global_prefetch_b8 on gfx1250

    BF16Vec a;
#pragma unroll
    for (int t = 0; t < 16; ++t) {
      float e = fs + fv.s[t];
      e = e > 0.f ? e : ALPHA_L * e;
      float p = (av.s[t] > 0) ? __expf(e - mi) * rs : 0.0f;
      a.s[t] = f2bf(p);
    }
#pragma unroll
    for (int n = 0; n < NT; ++n) {
      BF16Vec b;
      const unsigned short* bp =
          WhT + (size_t)(colHead + n * 16 + lo) * N_NODES + k0 + hi * 8;
      b.q[0] = *(const v4u*)(bp);
      b.q[1] = *(const v4u*)(bp + 16);
      acc[n] = __builtin_amdgcn_wmma_f32_16x16x32_bf16(
          false, a.v, false, b.v, (short)0, acc[n], false, false);
    }
  }

#pragma unroll
  for (int n = 0; n < NT; ++n)
#pragma unroll
    for (int r = 0; r < 8; ++r) {
      int row = Mbase + r + (hi << 3);
      int col = colHead + n * 16 + lo;
      float v = acc[n][r];
      if (DO_ELU) v = v > 0.f ? v : (__expf(v) - 1.0f);
      if (F32OUT) {
        outF[(size_t)row * (NT * 16) + col] = v;
      } else {
        unsigned short bv = f2bf(v);
        outBF[(size_t)row * ldOut + col] = bv;
        outBFT[(size_t)col * N_NODES + row] = bv;
      }
    }
}

// ---------------------------------------------------------------- launcher
extern "C" void kernel_launch(void* const* d_in, const int* in_sizes, int n_in,
                              void* d_out, int out_size, void* d_ws, size_t ws_size,
                              hipStream_t stream)
{
  (void)in_sizes; (void)n_in; (void)out_size; (void)ws_size;
  const float* x   = (const float*)d_in[0];
  const int*   adj = (const int*)  d_in[1];
  const float* W1  = (const float*)d_in[2];
  const float* a1  = (const float*)d_in[3];
  const float* W2  = (const float*)d_in[4];
  const float* a2  = (const float*)d_in[5];
  float* out = (float*)d_out;

  char* w = (char*)d_ws;
  auto alloc = [&](size_t bytes) -> void* {
    void* p = (void*)w;
    w += (bytes + 255) & ~(size_t)255;
    return p;
  };
  unsigned short* Xbf   = (unsigned short*)alloc((size_t)N_NODES * N_FEATC * 2);
  unsigned short* W1T   = (unsigned short*)alloc((size_t)N_FEATC * (N_HEADS * N_HID) * 2);
  unsigned short* W2T   = (unsigned short*)alloc((size_t)N_CLASS * N_FEATC * 2);
  float*          Wh    = (float*)         alloc((size_t)N_NODES * (N_HEADS * N_HID) * 4);
  unsigned short* WhT   = (unsigned short*)alloc((size_t)(N_HEADS * N_HID) * N_NODES * 2);
  float*          f1s   = (float*)         alloc((size_t)N_HEADS * N_NODES * 4);
  float*          f1d   = (float*)         alloc((size_t)N_HEADS * N_NODES * 4);
  float*          m1    = (float*)         alloc((size_t)N_HEADS * N_NODES * 4);
  float*          s1    = (float*)         alloc((size_t)N_HEADS * N_NODES * 4);
  unsigned short* hcat  = (unsigned short*)alloc((size_t)N_NODES * (N_HEADS * N_HID) * 2);
  unsigned short* hcatT = (unsigned short*)alloc((size_t)(N_HEADS * N_HID) * N_NODES * 2);
  float*          Wh2   = (float*)         alloc((size_t)N_NODES * N_CLASS * 4);
  unsigned short* Wh2T  = (unsigned short*)alloc((size_t)N_CLASS * N_NODES * 2);
  float*          f2s   = (float*)         alloc((size_t)N_NODES * 4);
  float*          f2d   = (float*)         alloc((size_t)N_NODES * 4);
  float*          m2    = (float*)         alloc((size_t)N_NODES * 4);
  float*          s2    = (float*)         alloc((size_t)N_NODES * 4);

  // 1. converts
  cvt_x_bf16<<<(N_NODES * N_FEATC + 255) / 256, 256, 0, stream>>>(
      x, Xbf, N_NODES * N_FEATC);
  cvt_w1_bf16<<<(N_HEADS * N_FEATC * N_HID + 255) / 256, 256, 0, stream>>>(W1, W1T);
  cvt_w2_bf16<<<(N_FEATC * N_CLASS + 255) / 256, 256, 0, stream>>>(W2, W2T);

  // 2. Wh = X @ W1cat  (4096 x 512 x 512)
  gemm_bf16<4><<<dim3(N_NODES / 128, (N_HEADS * N_HID) / 64), 256, 0, stream>>>(
      Xbf, W1T, Wh, WhT, N_HEADS * N_HID);

  // 3. attention logit vectors, 4. row stats
  fdot_kernel<N_HEADS, N_HID><<<(N_HEADS * N_NODES + 255) / 256, 256, 0, stream>>>(
      Wh, a1, f1s, f1d);
  row_stats<N_HEADS><<<N_NODES, 256, 0, stream>>>(adj, f1s, f1d, m1, s1);

  // 5. h_cat = ELU(att @ Wh), per head (8 x 4096 x 4096 x 64)
  att_mm<4, true, false><<<dim3(N_NODES / 128, N_HEADS), 256, 0, stream>>>(
      adj, f1s, f1d, m1, s1, WhT, hcat, hcatT, nullptr, N_HEADS * N_HID);

  // 6. Wh2 = h_cat @ W2 (4096 x 512 x 16)
  gemm_bf16<1><<<dim3(N_NODES / 128, 1), 256, 0, stream>>>(
      hcat, W2T, Wh2, Wh2T, N_CLASS);

  // 7/8. layer-2 logits + stats
  fdot_kernel<1, N_CLASS><<<(N_NODES + 255) / 256, 256, 0, stream>>>(
      Wh2, a2, f2s, f2d);
  row_stats<1><<<N_NODES, 256, 0, stream>>>(adj, f2s, f2d, m2, s2);

  // 9. out = att2 @ Wh2 (4096 x 4096 x 16), f32 result
  att_mm<1, false, true><<<dim3(N_NODES / 128, 1), 256, 0, stream>>>(
      adj, f2s, f2d, m2, s2, Wh2T, nullptr, nullptr, out, N_CLASS);
}
